// CPTMLP_86629490360409
// MI455X (gfx1250) — 8-bit matrix operand order repaired and checked by reading, compiled, not run on hardware
//
#include <hip/hip_runtime.h>
#include <hip/hip_bf16.h>
#include <stdint.h>


typedef __attribute__((ext_vector_type(8))) int v8i;
typedef __attribute__((ext_vector_type(4))) unsigned int u32x4;
typedef __attribute__((ext_vector_type(8))) int i32x8;
typedef __attribute__((ext_vector_type(4))) int i32x4;

#define TILE_M 128
#define TILE_N 128
#define TILE_K 64

#if defined(__HIP_DEVICE_COMPILE__) && __has_builtin(__builtin_amdgcn_tensor_load_to_lds)
#define USE_TDM 1
#else
#define USE_TDM 0
#endif

__device__ __forceinline__ void wait_tensorcnt0() {
#if __has_builtin(__builtin_amdgcn_s_wait_tensorcnt)
  __builtin_amdgcn_s_wait_tensorcnt(0);
#else
  asm volatile("s_wait_tensorcnt 0" ::: "memory");
#endif
}

#if USE_TDM
__device__ __forceinline__ unsigned lds_byte_off(const void* p) {
  return (unsigned)(uintptr_t)p;
}

__device__ __forceinline__ void tdm_load_tile_2d(unsigned lds_off,
                                                 const void* gtile,
                                                 unsigned tensor_d0,
                                                 unsigned tensor_d1,
                                                 unsigned tile_d0,
                                                 unsigned tile_d1,
                                                 unsigned long long stride0) {
  const unsigned long long ga = (unsigned long long)(uintptr_t)gtile;
  u32x4 g0;
  g0[0] = 1u;
  g0[1] = lds_off;
  g0[2] = (unsigned)(ga & 0xffffffffu);
  g0[3] = (unsigned)((ga >> 32) & 0x01ffffffu) | (2u << 30);

  i32x8 g1;
  g1[0] = 0;
  g1[1] = (int)((tensor_d0 & 0xffffu) << 16);
  g1[2] = (int)((tensor_d0 >> 16) | ((tensor_d1 & 0xffffu) << 16));
  g1[3] = (int)((tensor_d1 >> 16) | ((tile_d0 & 0xffffu) << 16));
  g1[4] = (int)(tile_d1 & 0xffffu);
  g1[5] = (int)(unsigned)(stride0 & 0xffffffffu);
  g1[6] = (int)(unsigned)((stride0 >> 32) & 0xffffu);
  g1[7] = 0;

  const i32x4 gz = {0, 0, 0, 0};
#if __clang_major__ >= 23
  const i32x8 gz8 = {0, 0, 0, 0, 0, 0, 0, 0};
  __builtin_amdgcn_tensor_load_to_lds(g0, g1, gz, gz, gz8, 0);
#else
  __builtin_amdgcn_tensor_load_to_lds(g0, g1, gz, gz, 0);
#endif
}
#endif

__global__ __launch_bounds__(256) void absmax_kernel(
    const float* __restrict__ x, size_t n, unsigned int* __restrict__ out) {
  __shared__ unsigned int sm;
  if (threadIdx.x == 0) sm = 0u;
  __syncthreads();
  float m = 0.0f;
  for (size_t i = (size_t)blockIdx.x * blockDim.x + threadIdx.x; i < n;
       i += (size_t)gridDim.x * blockDim.x)
    m = fmaxf(m, fabsf(x[i]));
  atomicMax(&sm, __float_as_uint(m));
  __syncthreads();
  if (threadIdx.x == 0) atomicMax(out, sm);
}

__global__ void init_absmax_kernel(unsigned int* __restrict__ p) {
  if (threadIdx.x < 4) p[threadIdx.x] = 0u;
}

__global__ void finalize_scale_kernel(const unsigned int* __restrict__ bits,
                                      float* __restrict__ scales) {
  int i = threadIdx.x;
  if (i < 4) scales[i] = fmaxf(__uint_as_float(bits[i]), 1e-8f) * (1.0f / 127.0f);
}

__global__ __launch_bounds__(256) void quant_kernel(
    const float* __restrict__ x, size_t n, const float* __restrict__ scale,
    signed char* __restrict__ q) {
  const float inv = 1.0f / scale[0];
  for (size_t i = (size_t)blockIdx.x * blockDim.x + threadIdx.x; i < n;
       i += (size_t)gridDim.x * blockDim.x) {
    float v = rintf(x[i] * inv);
    v = fminf(fmaxf(v, -127.0f), 127.0f);
    q[i] = (signed char)v;
  }
}

__global__ __launch_bounds__(256) void quant_transpose_kernel(
    const float* __restrict__ W, int K, int N, const float* __restrict__ scale,
    signed char* __restrict__ q) {
  __shared__ float tile[32][33];
  const int k0 = blockIdx.x * 32, n0 = blockIdx.y * 32;
  const int tx = threadIdx.x & 31, ty = threadIdx.x >> 5;
  const float inv = 1.0f / scale[0];
  for (int i = ty; i < 32; i += 8)
    tile[i][tx] = W[(size_t)(k0 + i) * N + (n0 + tx)];
  __syncthreads();
  for (int i = ty; i < 32; i += 8) {
    float v = rintf(tile[tx][i] * inv);
    v = fminf(fmaxf(v, -127.0f), 127.0f);
    q[(size_t)(n0 + i) * K + (k0 + tx)] = (signed char)v;
  }
}

__global__ __launch_bounds__(256) void lora_t_kernel(
    const float* __restrict__ x, const float* __restrict__ A, int K,
    float* __restrict__ t) {
  const int m = blockIdx.x;
  const int tid = threadIdx.x;
  float accl[16];
#pragma unroll
  for (int r = 0; r < 16; ++r) accl[r] = 0.0f;
  for (int k = tid; k < K; k += 256) {
    const float xv = x[(size_t)m * K + k];
#pragma unroll
    for (int r = 0; r < 16; ++r)
      accl[r] = fmaf(xv, A[(size_t)k * 16 + r], accl[r]);
  }
  __shared__ float red[256];
  for (int r = 0; r < 16; ++r) {
    red[tid] = accl[r];
    __syncthreads();
    for (int s = 128; s > 0; s >>= 1) {
      if (tid < s) red[tid] += red[tid + s];
      __syncthreads();
    }
    if (tid == 0) t[(size_t)m * 16 + r] = red[0];
    __syncthreads();
  }
}

__global__ __launch_bounds__(256) void gemm_q8_lora_kernel(
    const signed char* __restrict__ qA, const signed char* __restrict__ qB,
    const float* __restrict__ scaleA, const float* __restrict__ scaleB,
    const float* __restrict__ bias, const float* __restrict__ tlora,
    const float* __restrict__ Blora, float* __restrict__ out,
    unsigned int* __restrict__ absmax_out, int M, int N, int K,
    int apply_gelu) {
  __shared__ signed char As[2][TILE_M][TILE_K];
  __shared__ signed char Bs[2][TILE_N][TILE_K];
  __shared__ float tL[TILE_M][16];
  __shared__ float bL[16][TILE_N];
  __shared__ unsigned int blkmax;

  const int tid = threadIdx.x;
  const int lane = tid & 31;
  const int wave = tid >> 5;
  const int wm = wave & 3;
  const int wn = wave >> 2;
  const int bm = blockIdx.y * TILE_M;
  const int bn = blockIdx.x * TILE_N;

  for (int i = tid; i < TILE_M * 16; i += 256)
    tL[i >> 4][i & 15] = tlora[(size_t)(bm + (i >> 4)) * 16 + (i & 15)];
  for (int i = tid; i < 16 * TILE_N; i += 256) {
    const int r = i / TILE_N, c = i % TILE_N;
    bL[r][c] = Blora[(size_t)r * N + bn + c];
  }
  if (tid == 0) blkmax = 0u;

  const v8i vzero = {0, 0, 0, 0, 0, 0, 0, 0};
  v8i acc[2][4];
#pragma unroll
  for (int mf = 0; mf < 2; ++mf)
#pragma unroll
    for (int nf = 0; nf < 4; ++nf) acc[mf][nf] = vzero;

  const int ksteps = K / TILE_K;
  const int hiA = (lane < 16) ? 0 : 16;
  const int loB = (lane < 16) ? 0 : 16;

#if USE_TDM
  if (wave == 0) {
    tdm_load_tile_2d(lds_byte_off(&As[0][0][0]), qA + (size_t)bm * K,
                     (unsigned)K, (unsigned)M, TILE_K, TILE_M,
                     (unsigned long long)K);
    tdm_load_tile_2d(lds_byte_off(&Bs[0][0][0]), qB + (size_t)bn * K,
                     (unsigned)K, (unsigned)N, TILE_K, TILE_N,
                     (unsigned long long)K);
  }
#endif

  for (int kk = 0; kk < ksteps; ++kk) {
    const int cur = kk & 1;
    const int k0 = kk * TILE_K;

#if USE_TDM
    if (wave == 0) wait_tensorcnt0();
    __syncthreads();
    if (wave == 0 && kk + 1 < ksteps) {
      const int nxt = cur ^ 1;
      tdm_load_tile_2d(lds_byte_off(&As[nxt][0][0]),
                       qA + (size_t)bm * K + k0 + TILE_K, (unsigned)K,
                       (unsigned)M, TILE_K, TILE_M, (unsigned long long)K);
      tdm_load_tile_2d(lds_byte_off(&Bs[nxt][0][0]),
                       qB + (size_t)bn * K + k0 + TILE_K, (unsigned)K,
                       (unsigned)N, TILE_K, TILE_N, (unsigned long long)K);
    }
#else
#pragma unroll
    for (int c = 0; c < 2; ++c) {
      const int chunk = tid + c * 256;
      const int row = chunk >> 2;
      const int col = (chunk & 3) << 4;
      *(int4*)&As[cur][row][col] =
          *(const int4*)&qA[(size_t)(bm + row) * K + k0 + col];
      *(int4*)&Bs[cur][row][col] =
          *(const int4*)&qB[(size_t)(bn + row) * K + k0 + col];
    }
    if (kk + 1 < ksteps) {
      const int row = tid >> 1;
      __builtin_prefetch(&qA[(size_t)(bm + row) * K + k0 + TILE_K], 0, 3);
      __builtin_prefetch(&qB[(size_t)(bn + row) * K + k0 + TILE_K], 0, 3);
    }
    __syncthreads();
#endif

    v8i afrag[2];
#pragma unroll
    for (int mf = 0; mf < 2; ++mf) {
      const signed char* ap = &As[cur][wm * 32 + mf * 16 + (lane & 15)][0];
      const int2 a0 = *(const int2*)(ap + hiA + 0);
      const int2 a1 = *(const int2*)(ap + hiA + 8);
      const int2 a2 = *(const int2*)(ap + hiA + 32);
      const int2 a3 = *(const int2*)(ap + hiA + 40);
      v8i a;
      a[0] = a0.x; a[1] = a0.y; a[2] = a1.x; a[3] = a1.y;
      a[4] = a2.x; a[5] = a2.y; a[6] = a3.x; a[7] = a3.y;
      afrag[mf] = a;
    }
    v8i bfrag[4];
#pragma unroll
    for (int nf = 0; nf < 4; ++nf) {
      const signed char* bp = &Bs[cur][wn * 64 + nf * 16 + (lane & 15)][0];
      const int4 b0 = *(const int4*)(bp + loB);
      const int4 b1 = *(const int4*)(bp + loB + 32);
      v8i b;
      b[0] = b0.x; b[1] = b0.y; b[2] = b0.z; b[3] = b0.w;
      b[4] = b1.x; b[5] = b1.y; b[6] = b1.z; b[7] = b1.w;
      bfrag[nf] = b;
    }

#pragma unroll
    for (int mf = 0; mf < 2; ++mf)
#pragma unroll
      for (int nf = 0; nf < 4; ++nf)
        acc[mf][nf] = __builtin_amdgcn_wmma_i32_16x16x64_iu8(
             true, afrag[mf],  true, bfrag[nf],
            acc[mf][nf],  false,  false);

    __syncthreads();
  }

  const float sAB = scaleA[0] * scaleB[0];
  float lmax = 0.0f;
#pragma unroll
  for (int mf = 0; mf < 2; ++mf) {
#pragma unroll
    for (int nf = 0; nf < 4; ++nf) {
      const int nloc = wn * 64 + nf * 16 + (lane & 15);
      const float bv = bias[bn + nloc];
#pragma unroll
      for (int i = 0; i < 8; ++i) {
        const int mloc = wm * 32 + mf * 16 + i + ((lane < 16) ? 0 : 8);
        float v = (float)acc[mf][nf][i] * sAB + bv;
        float lora = 0.0f;
#pragma unroll
        for (int r = 0; r < 16; ++r)
          lora = fmaf(tL[mloc][r], bL[r][nloc], lora);
        v += 2.0f * lora;
        if (apply_gelu)
          v = 0.5f * v * (1.0f + erff(v * 0.70710678118654752440f));
        out[(size_t)(bm + mloc) * N + (bn + nloc)] = v;
        lmax = fmaxf(lmax, fabsf(v));
      }
    }
  }

  if (absmax_out) {
    atomicMax(&blkmax, __float_as_uint(lmax));
    __syncthreads();
    if (tid == 0) atomicMax(absmax_out, blkmax);
  }
}

extern "C" void kernel_launch(void* const* d_in, const int* in_sizes, int n_in,
                              void* d_out, int out_size, void* d_ws,
                              size_t ws_size, hipStream_t stream) {
  const float* x = (const float*)d_in[0];
  const float* W_fc = (const float*)d_in[1];
  const float* b_fc = (const float*)d_in[2];
  const float* A_fc = (const float*)d_in[3];
  const float* B_fc = (const float*)d_in[4];
  const float* W_proj = (const float*)d_in[5];
  const float* b_proj = (const float*)d_in[6];
  const float* A_proj = (const float*)d_in[7];
  const float* B_proj = (const float*)d_in[8];
  float* out = (float*)d_out;

  const int d = 2048, dff = 8192, r = 16;
  const int M = in_sizes[0] / d;

  char* ws = (char*)d_ws;
  unsigned int* u_absmax = (unsigned int*)ws;
  float* scales = (float*)(ws + 16);
  size_t off = 256;
  signed char* qx = (signed char*)(ws + off);  off += (size_t)M * d;
  signed char* qW1 = (signed char*)(ws + off); off += (size_t)d * dff;
  signed char* qh = (signed char*)(ws + off);  off += (size_t)M * dff;
  signed char* qW2 = (signed char*)(ws + off); off += (size_t)dff * d;
  float* h = (float*)(ws + off);               off += (size_t)M * dff * sizeof(float);
  float* t1 = (float*)(ws + off);              off += (size_t)M * r * sizeof(float);
  float* t2 = (float*)(ws + off);              off += (size_t)M * r * sizeof(float);
  (void)n_in; (void)out_size; (void)ws_size;

  init_absmax_kernel<<<1, 64, 0, stream>>>(u_absmax);

  absmax_kernel<<<1024, 256, 0, stream>>>(x, (size_t)M * d, &u_absmax[0]);
  absmax_kernel<<<1024, 256, 0, stream>>>(W_fc, (size_t)d * dff, &u_absmax[1]);
  absmax_kernel<<<1024, 256, 0, stream>>>(W_proj, (size_t)dff * d, &u_absmax[3]);
  finalize_scale_kernel<<<1, 4, 0, stream>>>(u_absmax, scales);

  quant_kernel<<<2048, 256, 0, stream>>>(x, (size_t)M * d, &scales[0], qx);
  quant_transpose_kernel<<<dim3(d / 32, dff / 32), 256, 0, stream>>>(
      W_fc, d, dff, &scales[1], qW1);
  quant_transpose_kernel<<<dim3(dff / 32, d / 32), 256, 0, stream>>>(
      W_proj, dff, d, &scales[3], qW2);
  lora_t_kernel<<<M, 256, 0, stream>>>(x, A_fc, d, t1);

  gemm_q8_lora_kernel<<<dim3(dff / TILE_N, M / TILE_M), 256, 0, stream>>>(
      qx, qW1, &scales[0], &scales[1], b_fc, t1, B_fc, h, &u_absmax[2],
      M, dff, d,  1);

  finalize_scale_kernel<<<1, 4, 0, stream>>>(u_absmax, scales);
  quant_kernel<<<4096, 256, 0, stream>>>(h, (size_t)M * dff, &scales[2], qh);
  lora_t_kernel<<<M, 256, 0, stream>>>(h, A_proj, dff, t2);

  gemm_q8_lora_kernel<<<dim3(d / TILE_N, M / TILE_M), 256, 0, stream>>>(
      qh, qW2, &scales[2], &scales[3], b_proj, t2, B_proj, out, nullptr,
      M, d, dff,  0);
}
